// MultiHeadAttention_39213051412948
// MI455X (gfx1250) — compile-verified, matrix operand order checked by reading, not run on hardware
//
#include <hip/hip_runtime.h>
#include <hip/hip_bf16.h>

#define BB 2
#define SS 2048
#define DM 1024
#define NH 16
#define HD 64
#define MTOT (BB*SS)  // 4096

typedef __bf16 bf16_t;
typedef bf16_t v16bf __attribute__((ext_vector_type(16)));
typedef float  v8f   __attribute__((ext_vector_type(8)));

// ---------------------------------------------------------------------------
// Fragment loader: 16x32 tile of a K-contiguous bf16 matrix, element (row,k)
// at base[row*ld + k]. Matches the CDNA5 16-bit A layout:
//   lanes 0-15  : row = lane,     K = {0..7, 16..23}
//   lanes 16-31 : row = lane-16,  K = {8..15, 24..31}
// Works for both global and LDS pointers (compiler picks b128 loads).
// ---------------------------------------------------------------------------
static __device__ __forceinline__ v16bf load_frag(const bf16_t* base, int ld) {
  const int l   = threadIdx.x & 31;
  const int row = l & 15;
  const int kh  = (l >> 4) << 3;       // 0 or 8
  const unsigned* p32 = (const unsigned*)(base + row * ld + kh);
  union { v16bf v; unsigned u[8]; } f;
#pragma unroll
  for (int i = 0; i < 4; ++i) {
    f.u[i]     = p32[i];       // K = kh + 2i, 2i+1
    f.u[i + 4] = p32[i + 8];   // K = kh + 16 + 2i, +1
  }
  return f.v;
}

static __device__ __forceinline__ v8f wmma_bf16(v16bf a, v16bf b, v8f c) {
  return __builtin_amdgcn_wmma_f32_16x16x32_bf16(
      /*neg_a=*/false, a, /*neg_b=*/false, b,
      /*c_mod=*/(short)0, c, /*reuse_a=*/false, /*reuse_b=*/false);
}

// Async global -> LDS copy, 16 bytes per lane (CDNA5 ASYNCcnt path).
static __device__ __forceinline__ void async_copy_b128(const bf16_t* g, const bf16_t* lds_generic) {
  unsigned ldsoff = (unsigned)(size_t)(__attribute__((address_space(3))) const bf16_t*)lds_generic;
  unsigned long long ga = (unsigned long long)(size_t)g;
  asm volatile("global_load_async_to_lds_b128 %0, %1, off"
               :: "v"(ldsoff), "v"(ga) : "memory");
}

static __device__ __forceinline__ void wait_async0() {
#if __has_builtin(__builtin_amdgcn_s_wait_asynccnt)
  __builtin_amdgcn_s_wait_asynccnt(0);
#else
  asm volatile("s_wait_asynccnt 0x0" ::: "memory");
#endif
}

// ---------------------------------------------------------------------------
// 1) fp32 -> bf16 convert (x), and transpose-convert weights [K][N] -> [N][K]
// ---------------------------------------------------------------------------
__global__ void cvt_x_kernel(const float* __restrict__ x, bf16_t* __restrict__ xb, int n) {
  int i = blockIdx.x * blockDim.x + threadIdx.x;
  if (i < n) xb[i] = (bf16_t)x[i];
}

__global__ void transpose_w_kernel(const float* __restrict__ w, bf16_t* __restrict__ wt) {
  int i = blockIdx.x * blockDim.x + threadIdx.x;   // over 1024*1024
  int n = i & (DM - 1);
  int k = i >> 10;
  wt[(size_t)n * DM + k] = (bf16_t)w[i];
}

// ---------------------------------------------------------------------------
// 2) Fused QKV GEMM, register-blocked: each wave computes a 32x64 tile
//    (2 A-frags + 4 B-frags -> 8 WMMAs per 32-deep K step).
//    grid = (M/32, N/256, 3), block = 128 (4 waves).
//    Q,K stored [B,H,S,HD] bf16; V stored transposed [B,H,HD,S] bf16.
// ---------------------------------------------------------------------------
__global__ void __launch_bounds__(128)
qkv_gemm_kernel(const bf16_t* __restrict__ xb,
                const bf16_t* __restrict__ wqt, const bf16_t* __restrict__ wkt,
                const bf16_t* __restrict__ wvt,
                const float* __restrict__ bq, const float* __restrict__ bk,
                const float* __restrict__ bv,
                bf16_t* __restrict__ Qb, bf16_t* __restrict__ Kb,
                bf16_t* __restrict__ Vt)
{
  const int wave  = threadIdx.x >> 5;
  const int m0    = blockIdx.x * 32;
  const int n0    = (blockIdx.y * 4 + wave) * 64;
  const int which = blockIdx.z;

  const bf16_t* wt   = (which == 0) ? wqt : (which == 1) ? wkt : wvt;
  const float*  bias = (which == 0) ? bq  : (which == 1) ? bk  : bv;

  v8f acc[2][4] = {};
  for (int k0 = 0; k0 < DM; k0 += 32) {
    v16bf a0 = load_frag(xb + (size_t)m0 * DM + k0, DM);
    v16bf a1 = load_frag(xb + (size_t)(m0 + 16) * DM + k0, DM);
#pragma unroll
    for (int j = 0; j < 4; ++j) {
      v16bf b = load_frag(wt + (size_t)(n0 + 16 * j) * DM + k0, DM);
      acc[0][j] = wmma_bf16(a0, b, acc[0][j]);
      acc[1][j] = wmma_bf16(a1, b, acc[1][j]);
    }
  }

  const int l     = threadIdx.x & 31;
  const int rbase = (l >> 4) << 3;

#pragma unroll
  for (int j = 0; j < 4; ++j) {
    const int n  = n0 + 16 * j + (l & 15);
    const float bn = bias[n];
    const int h = n >> 6, d = n & 63;
#pragma unroll
    for (int i = 0; i < 2; ++i) {
#pragma unroll
      for (int r = 0; r < 8; ++r) {
        int m  = m0 + 16 * i + rbase + r;
        int b_ = m >> 11;          // / SS
        int s  = m & (SS - 1);
        float v = acc[i][j][r] + bn;
        if (which == 2) {
          Vt[(((size_t)(b_ * NH + h)) * HD + d) * SS + s] = (bf16_t)v;
        } else {
          bf16_t* dst = (which == 0) ? Qb : Kb;
          dst[(((size_t)(b_ * NH + h)) * SS + s) * HD + d] = (bf16_t)v;
        }
      }
    }
  }
}

// ---------------------------------------------------------------------------
// 3) RoPE over [B,H,S,HD] bf16, one thread per (bh, s, pair j).
// ---------------------------------------------------------------------------
__global__ void rope_kernel(bf16_t* __restrict__ p) {
  int idx = blockIdx.x * blockDim.x + threadIdx.x;
  if (idx >= BB * NH * SS * (HD / 2)) return;
  int j  = idx & 31;
  int s  = (idx >> 5) & (SS - 1);
  int bh = idx >> 16;
  bf16_t* q = p + ((size_t)bh * SS + s) * HD + 2 * j;
  float e = (float)q[0], o = (float)q[1];
  float inv = __expf(-((float)(2 * j) / (float)HD) * 9.2103403719761836f);
  float ang = (float)s * inv;
  float sn, c;
  __sincosf(ang, &sn, &c);
  q[0] = (bf16_t)(e * c - o * sn);
  q[1] = (bf16_t)(e * sn + o * c);
}

// ---------------------------------------------------------------------------
// 4) Flash attention with cooperative async-to-LDS staging.
//    Block = 4 waves sharing one (b,h) and 64 consecutive queries.
//    Per 32-key step: the block stages K (32x64) and V^T (64x32) tiles into
//    LDS with global_load_async_to_lds_b128, fences with s_wait_asynccnt +
//    barrier, then each wave runs score WMMAs, online softmax, and PV WMMAs
//    reading fragments from LDS. Causal divergence is predicated compute
//    under uniform barriers (all waves loop to the block-max key tile).
// ---------------------------------------------------------------------------
__global__ void __launch_bounds__(128)
flash_kernel(const bf16_t* __restrict__ Qb, const bf16_t* __restrict__ Kb,
             const bf16_t* __restrict__ Vt, bf16_t* __restrict__ Ob)
{
  __shared__ bf16_t kTile[32 * 64];     // [key][dim]  4KB
  __shared__ bf16_t vTile[64 * 32];     // [dim][key]  4KB
  __shared__ bf16_t pTile[4][16 * 32];  // per-wave P  4KB

  const int wave = threadIdx.x >> 5;
  const int l    = threadIdx.x & 31;
  const int t    = threadIdx.x;         // 0..127
  const int tile = blockIdx.x * 4 + wave;
  const int qt   = tile & 127;          // S/16 = 128 query tiles
  const int bh   = tile >> 7;           // uniform across block (4 | 128)
  const int q0   = qt * 16;
  const int qt0  = (blockIdx.x * 4) & 127;

  const bf16_t* Qbase = Qb + ((size_t)bh * SS + q0) * HD;
  const bf16_t* Kbh   = Kb + (size_t)bh * SS * HD;
  const bf16_t* Vbh   = Vt + (size_t)bh * HD * SS;

  // Q fragments for K-dim 0..31 and 32..63, reused across all key tiles.
  v16bf aq0 = load_frag(Qbase + 0, HD);
  v16bf aq1 = load_frag(Qbase + 32, HD);

  v8f o0 = {}, o1 = {}, o2 = {}, o3 = {};
  float mrow[8], lrow[8];
#pragma unroll
  for (int r = 0; r < 8; ++r) { mrow[r] = -1e30f; lrow[r] = 0.0f; }

  const int row      = l & 15;
  const int rbase    = (l >> 4) << 3;
  const int my_ktmax = (q0 + 15) >> 5;
  const int bk_ktmax = (qt0 * 16 + 63) >> 5;   // uniform per block
  bf16_t* myp = pTile[wave];

  for (int kt = 0; kt <= bk_ktmax; ++kt) {
    const int j0 = kt << 5;

    __syncthreads();   // previous iteration's LDS reads complete

    // Cooperative staging: 128 threads x 2 chunks x 16B per tile.
#pragma unroll
    for (int i = 0; i < 2; ++i) {
      int flat = i * 128 + t;             // 0..255
      // K tile: key = flat/8, dims 8*(flat%8) .. +7
      int key = flat >> 3, dg = (flat & 7) << 3;
      async_copy_b128(Kbh + (size_t)(j0 + key) * HD + dg, kTile + key * HD + dg);
      // V tile: dim = flat/4, keys 8*(flat%4) .. +7
      int dim = flat >> 2, kg = (flat & 3) << 3;
      async_copy_b128(Vbh + (size_t)dim * SS + j0 + kg, vTile + dim * 32 + kg);
    }
    wait_async0();
    __syncthreads();   // all waves' tiles visible

    if (kt <= my_ktmax) {
      // S = Q @ K^T from LDS K tile (ld = 64)
      v8f s0 = {}, s1 = {};
      s0 = wmma_bf16(aq0, load_frag(kTile + 0,            HD), s0);
      s0 = wmma_bf16(aq1, load_frag(kTile + 32,           HD), s0);
      s1 = wmma_bf16(aq0, load_frag(kTile + 16 * HD,      HD), s1);
      s1 = wmma_bf16(aq1, load_frag(kTile + 16 * HD + 32, HD), s1);

      const bool partial = (j0 + 31) > q0;
#pragma unroll
      for (int r = 0; r < 8; ++r) {
        float x0 = s0[r] * 0.125f;     // 1/sqrt(64)
        float x1 = s1[r] * 0.125f;
        if (partial) {
          int q = q0 + rbase + r;
          if (j0 + row > q)      x0 = -1e30f;
          if (j0 + 16 + row > q) x1 = -1e30f;
        }
        s0[r] = x0; s1[r] = x1;
      }

      // Online softmax per row (row lives across a 16-lane group).
#pragma unroll
      for (int r = 0; r < 8; ++r) {
        float mx = fmaxf(s0[r], s1[r]);
        for (int msk = 8; msk; msk >>= 1) mx = fmaxf(mx, __shfl_xor(mx, msk, 16));
        float mnew = fmaxf(mrow[r], mx);
        float p0 = __expf(s0[r] - mnew);
        float p1 = __expf(s1[r] - mnew);
        float sum = p0 + p1;
        for (int msk = 8; msk; msk >>= 1) sum += __shfl_xor(sum, msk, 16);
        float alpha = __expf(mrow[r] - mnew);
        lrow[r] = lrow[r] * alpha + sum;
        mrow[r] = mnew;
        o0[r] *= alpha; o1[r] *= alpha; o2[r] *= alpha; o3[r] *= alpha;
        myp[(rbase + r) * 32 + row]      = (bf16_t)p0;
        myp[(rbase + r) * 32 + row + 16] = (bf16_t)p1;
      }

      // Same-wave LDS store->load ordering, then P in A-layout.
      asm volatile("s_wait_dscnt 0x0" ::: "memory");
      v16bf ap = load_frag(myp, 32);

      // O += P @ V from LDS V tile (ld = 32)
      o0 = wmma_bf16(ap, load_frag(vTile +  0 * 32, 32), o0);
      o1 = wmma_bf16(ap, load_frag(vTile + 16 * 32, 32), o1);
      o2 = wmma_bf16(ap, load_frag(vTile + 32 * 32, 32), o2);
      o3 = wmma_bf16(ap, load_frag(vTile + 48 * 32, 32), o3);
    }
  }

  // Normalize, write attention output as bf16 to Ob in [B,S,D] layout.
  const int b_ = bh >> 4, h = bh & 15;
#pragma unroll
  for (int r = 0; r < 8; ++r) {
    int s = q0 + rbase + r;
    float inv = 1.0f / lrow[r];
    size_t base = ((size_t)b_ * SS + s) * DM + h * HD;
    Ob[base +  0 + row] = (bf16_t)(o0[r] * inv);
    Ob[base + 16 + row] = (bf16_t)(o1[r] * inv);
    Ob[base + 32 + row] = (bf16_t)(o2[r] * inv);
    Ob[base + 48 + row] = (bf16_t)(o3[r] * inv);
  }
}

// ---------------------------------------------------------------------------
// 5) Output projection: out = Ob @ wo + bo, fp32 result, 32x64 per wave.
// ---------------------------------------------------------------------------
__global__ void __launch_bounds__(128)
proj_kernel(const bf16_t* __restrict__ Ob, const bf16_t* __restrict__ wot,
            const float* __restrict__ bo, float* __restrict__ out)
{
  const int wave = threadIdx.x >> 5;
  const int m0 = blockIdx.x * 32;
  const int n0 = (blockIdx.y * 4 + wave) * 64;

  v8f acc[2][4] = {};
  for (int k0 = 0; k0 < DM; k0 += 32) {
    v16bf a0 = load_frag(Ob + (size_t)m0 * DM + k0, DM);
    v16bf a1 = load_frag(Ob + (size_t)(m0 + 16) * DM + k0, DM);
#pragma unroll
    for (int j = 0; j < 4; ++j) {
      v16bf b = load_frag(wot + (size_t)(n0 + 16 * j) * DM + k0, DM);
      acc[0][j] = wmma_bf16(a0, b, acc[0][j]);
      acc[1][j] = wmma_bf16(a1, b, acc[1][j]);
    }
  }

  const int l = threadIdx.x & 31;
  const int rbase = (l >> 4) << 3;
#pragma unroll
  for (int j = 0; j < 4; ++j) {
    const int n = n0 + 16 * j + (l & 15);
    const float bn = bo[n];
#pragma unroll
    for (int i = 0; i < 2; ++i)
#pragma unroll
      for (int r = 0; r < 8; ++r)
        out[(size_t)(m0 + 16 * i + rbase + r) * DM + n] = acc[i][j][r] + bn;
  }
}

// ---------------------------------------------------------------------------
extern "C" void kernel_launch(void* const* d_in, const int* in_sizes, int n_in,
                              void* d_out, int out_size, void* d_ws, size_t ws_size,
                              hipStream_t stream) {
  (void)in_sizes; (void)n_in; (void)out_size; (void)ws_size;
  const float* x  = (const float*)d_in[0];
  const float* wq = (const float*)d_in[1];
  const float* bq = (const float*)d_in[2];
  const float* wk = (const float*)d_in[3];
  const float* bk = (const float*)d_in[4];
  const float* wv = (const float*)d_in[5];
  const float* bv = (const float*)d_in[6];
  const float* wo = (const float*)d_in[7];
  const float* bo = (const float*)d_in[8];
  float* out = (float*)d_out;

  // Workspace carve-up (bf16 elements): 24M elems = 48 MB total.
  bf16_t* ws  = (bf16_t*)d_ws;
  bf16_t* xb  = ws;                              // 4M  : x bf16 [M,K]
  bf16_t* wqt = xb  + (size_t)MTOT * DM;         // 1M  : wq^T [N,K]
  bf16_t* wkt = wqt + (size_t)DM * DM;           // 1M
  bf16_t* wvt = wkt + (size_t)DM * DM;           // 1M
  bf16_t* wot = wvt + (size_t)DM * DM;           // 1M
  bf16_t* Qb  = wot + (size_t)DM * DM;           // 4M  : [B,H,S,HD]
  bf16_t* Kb  = Qb  + (size_t)MTOT * DM;         // 4M  : [B,H,S,HD]
  bf16_t* Vt  = Kb  + (size_t)MTOT * DM;         // 4M  : [B,H,HD,S]
  bf16_t* Ob  = Vt  + (size_t)MTOT * DM;         // 4M  : [B,S,D]

  const int NX = MTOT * DM;        // 4,194,304
  const int NW = DM * DM;          // 1,048,576

  cvt_x_kernel<<<NX / 256, 256, 0, stream>>>(x, xb, NX);
  transpose_w_kernel<<<NW / 256, 256, 0, stream>>>(wq, wqt);
  transpose_w_kernel<<<NW / 256, 256, 0, stream>>>(wk, wkt);
  transpose_w_kernel<<<NW / 256, 256, 0, stream>>>(wv, wvt);
  transpose_w_kernel<<<NW / 256, 256, 0, stream>>>(wo, wot);

  qkv_gemm_kernel<<<dim3(MTOT / 32, DM / 256, 3), 128, 0, stream>>>(
      xb, wqt, wkt, wvt, bq, bk, bv, Qb, Kb, Vt);

  const int NR = BB * NH * SS * (HD / 2);  // 2,097,152
  rope_kernel<<<NR / 256, 256, 0, stream>>>(Qb);
  rope_kernel<<<NR / 256, 256, 0, stream>>>(Kb);

  flash_kernel<<<(BB * NH * (SS / 16)) / 4, 128, 0, stream>>>(Qb, Kb, Vt, Ob);

  proj_kernel<<<dim3(MTOT / 32, DM / 256), 128, 0, stream>>>(Ob, wot, bo, out);
}
